// PrototypeLoss_49048526520788
// MI455X (gfx1250) — matrix operand order repaired and checked by reading, not compiled, not run on hardware
//
#include <hip/hip_runtime.h>
#include <hip/hip_bf16.h>

typedef __attribute__((ext_vector_type(16))) _Float16 v16h;
typedef __attribute__((ext_vector_type(8)))  _Float16 v8h;
typedef __attribute__((ext_vector_type(4)))  _Float16 v4h;
typedef __attribute__((ext_vector_type(8)))  float    v8f;
typedef __attribute__((ext_vector_type(4)))  int      v4i;

#define D_DIM 128
#define C_DIM 64
#define CHUNK 256
#define ZSTRIDE 136            // 128 + 8 halves pad: 272B row -> bank stride 4, 16B aligned
#define NRM_EPS 1e-12f

// workspace layout (floats): [0] ce_sum ; [1..64] counts ; [65..128] sq ; [129..) sums[64][128]
#define WS_CE   0
#define WS_CNT  1
#define WS_SQ   (1 + C_DIM)
#define WS_SUMS (1 + 2 * C_DIM)

// LDS layout (bytes), all 16B aligned
#define L_ZT    0                              // CHUNK * ZSTRIDE f16 = 69632
#define L_ATAB  (CHUNK * ZSTRIDE * 2)          // 8ks*4t*32lane*16h f16 = 32768
#define L_SLAB  (L_ATAB + 32768)               // CHUNK int = 1024
#define L_SINV  (L_SLAB + CHUNK * 4)           // CHUNK f16 = 512
#define L_SCNT  (L_SINV + CHUNK * 2)           // 64 f32
#define L_SSQ   (L_SCNT + C_DIM * 4)           // 64 f32
#define L_TOTAL (L_SSQ + C_DIM * 4)            // = 104448

// ---------------- Cross-entropy: one pass over logits ----------------
__global__ void __launch_bounds__(256) ce_kernel(const float* __restrict__ logits,
                                                 const int* __restrict__ labels,
                                                 float* __restrict__ ws, int N) {
  int tid = blockIdx.x * blockDim.x + threadIdx.x;
  int stride = gridDim.x * blockDim.x;
  float local = 0.0f;
  for (int r = tid; r < N; r += stride) {
    const float4* p = reinterpret_cast<const float4*>(logits + (size_t)r * C_DIM);
    float4 v[16];
    float m = -1e30f;
#pragma unroll
    for (int j = 0; j < 16; ++j) {
      v[j] = p[j];
      m = fmaxf(m, fmaxf(fmaxf(v[j].x, v[j].y), fmaxf(v[j].z, v[j].w)));
    }
    float s = 0.0f;
#pragma unroll
    for (int j = 0; j < 16; ++j) {
      s += __expf(v[j].x - m) + __expf(v[j].y - m) +
           __expf(v[j].z - m) + __expf(v[j].w - m);
    }
    int lab = labels[r];
    float tval = logits[(size_t)r * C_DIM + lab];
    local += (m + __logf(s)) - tval;
  }
#pragma unroll
  for (int off = 16; off > 0; off >>= 1) local += __shfl_down(local, off, 32);
  if ((threadIdx.x & 31) == 0) atomicAdd(&ws[WS_CE], local);
}

// ---------------- Proto stats: f16 stage + shared A-table + WMMA ----------------
__global__ void __launch_bounds__(256) proto_kernel(const float* __restrict__ emb,
                                                    const int* __restrict__ labels,
                                                    float* __restrict__ ws, int N) {
  extern __shared__ char smem[];
  _Float16* zt   = (_Float16*)(smem + L_ZT);
  _Float16* atab = (_Float16*)(smem + L_ATAB);
  int*      slab = (int*)     (smem + L_SLAB);
  _Float16* sinv = (_Float16*)(smem + L_SINV);
  float*    scnt = (float*)   (smem + L_SCNT);
  float*    ssq  = (float*)   (smem + L_SSQ);

  const int tid  = threadIdx.x;
  const int lane = tid & 31;
  const int wave = tid >> 5;
  const int laneHi = lane >> 4;
  const int d0   = wave * 16;           // this wave's D slice

  if (tid < C_DIM) { scnt[tid] = 0.0f; ssq[tid] = 0.0f; }
  __syncthreads();

  v8f acc[4] = {v8f{}, v8f{}, v8f{}, v8f{}};

  const int numChunks = N / CHUNK;
  for (int ch = blockIdx.x; ch < numChunks; ch += gridDim.x) {
    const int base = ch * CHUNK;

    // ---- phase 1: stage raw f16 row + norm + label + histogram ----
    {
      const int r = base + tid;
      const float4* p = reinterpret_cast<const float4*>(emb + (size_t)r * D_DIM);
      _Float16* zrow = zt + (size_t)tid * ZSTRIDE;
      float ss = 0.0f;
#pragma unroll
      for (int j = 0; j < 32; ++j) {
        float4 e = p[j];
        ss = fmaf(e.x, e.x, fmaf(e.y, e.y, fmaf(e.z, e.z, fmaf(e.w, e.w, ss))));
        v4h t;
        t[0] = (_Float16)e.x; t[1] = (_Float16)e.y;
        t[2] = (_Float16)e.z; t[3] = (_Float16)e.w;
        *(v4h*)(zrow + 4 * j) = t;     // unscaled: scaling folded into A matrix
      }
      float inv = 1.0f / fmaxf(sqrtf(ss), NRM_EPS);
      int lab = labels[r];
      slab[tid] = lab;
      sinv[tid] = (_Float16)inv;
      atomicAdd(&scnt[lab], 1.0f);
      atomicAdd(&ssq[lab], ss * inv * inv);
    }
    __syncthreads();

    // ---- phase 2: build shared A-fragment table (scaled one-hot) ----
    // slot = (ks*4 + tile)*32 + lane ; value a[h] = (label_k==cls) ? inv_k : 0
#pragma unroll
    for (int q = 0; q < 4; ++q) {
      const int slot   = tid + 256 * q;        // 0..1023
      const int lane_s = slot & 31;
      const int tt     = (slot >> 5) & 3;
      const int ks     = slot >> 7;
      const int lhi    = lane_s >> 4;
      const int m      = lane_s & 15;
      const int kb     = ks * 32 + 8 * lhi;    // K runs: kb..kb+7, kb+16..kb+23
      v4i l0 = *(const v4i*)(slab + kb);
      v4i l1 = *(const v4i*)(slab + kb + 4);
      v4i l2 = *(const v4i*)(slab + kb + 16);
      v4i l3 = *(const v4i*)(slab + kb + 20);
      v8h i0 = *(const v8h*)(sinv + kb);
      v8h i1 = *(const v8h*)(sinv + kb + 16);
      const int cls = tt * 16 + m;
      v16h a;
#pragma unroll
      for (int h = 0; h < 4; ++h) {
        a[h]      = (l0[h] == cls) ? i0[h]     : (_Float16)0.0f;
        a[h + 4]  = (l1[h] == cls) ? i0[h + 4] : (_Float16)0.0f;
        a[h + 8]  = (l2[h] == cls) ? i1[h]     : (_Float16)0.0f;
        a[h + 12] = (l3[h] == cls) ? i1[h + 4] : (_Float16)0.0f;
      }
      *(v16h*)(atab + (size_t)slot * 16) = a;  // 32B aligned
    }
    __syncthreads();

    // ---- phase 3: pure DS-load + WMMA ----
    for (int ks = 0; ks < CHUNK / 32; ++ks) {
      const int koff = ks * 32;
      const _Float16* zp = zt + (size_t)(koff + 8 * laneHi) * ZSTRIDE + d0 + (lane & 15);
      v16h b;
#pragma unroll
      for (int s = 0; s < 8; ++s) {
        b[s]     = zp[(size_t)s * ZSTRIDE];
        b[s + 8] = zp[(size_t)(s + 16) * ZSTRIDE];
      }
      const _Float16* ap = atab + (size_t)((ks * 4) * 32 + lane) * 16;
      v16h a0 = *(const v16h*)(ap);
      v16h a1 = *(const v16h*)(ap + 32 * 16);
      v16h a2 = *(const v16h*)(ap + 64 * 16);
      v16h a3 = *(const v16h*)(ap + 96 * 16);
      acc[0] = __builtin_amdgcn_wmma_f32_16x16x32_f16(false, a0, false, b, (short)0, acc[0], false, false);
      acc[1] = __builtin_amdgcn_wmma_f32_16x16x32_f16(false, a1, false, b, (short)0, acc[1], false, false);
      acc[2] = __builtin_amdgcn_wmma_f32_16x16x32_f16(false, a2, false, b, (short)0, acc[2], false, false);
      acc[3] = __builtin_amdgcn_wmma_f32_16x16x32_f16(false, a3, false, b, (short)0, acc[3], false, false);
    }
    __syncthreads();   // protect LDS before next chunk's staging
  }

  // ---- flush register accumulators: C/D layout -> (class, dim) ----
  const int dim = d0 + (lane & 15);
#pragma unroll
  for (int t = 0; t < 4; ++t)
#pragma unroll
    for (int g = 0; g < 8; ++g) {
      int cls = t * 16 + g + 8 * laneHi;
      atomicAdd(&ws[WS_SUMS + cls * D_DIM + dim], acc[t][g]);
    }
  __syncthreads();
  if (tid < C_DIM) {
    atomicAdd(&ws[WS_CNT + tid], scnt[tid]);
    atomicAdd(&ws[WS_SQ + tid],  ssq[tid]);
  }
}

// ---------------- Finalize: scalar combine ----------------
__global__ void finalize_kernel(const float* __restrict__ ws, float* __restrict__ out, int N) {
  __shared__ float s_per[C_DIM];
  __shared__ float s_val[C_DIM];
  const int c = threadIdx.x;
  if (c < C_DIM) {
    float cnt  = ws[WS_CNT + c];
    float safe = fmaxf(cnt, 1.0f);
    const float* srow = ws + WS_SUMS + c * D_DIM;
    float musq = 0.0f;
    for (int d = 0; d < D_DIM; ++d) {
      float mu = srow[d] / safe;
      musq += mu * mu;
    }
    float ssd = ws[WS_SQ + c] - cnt * musq;
    float per = ssd / safe;
    bool valid = cnt > 1.0f;
    s_per[c] = valid ? per : 0.0f;
    s_val[c] = valid ? 1.0f : 0.0f;
  }
  __syncthreads();
  if (c == 0) {
    float sp = 0.0f, sv = 0.0f;
    for (int i = 0; i < C_DIM; ++i) { sp += s_per[i]; sv += s_val[i]; }
    float proto = sp / fmaxf(sv, 1.0f);
    float ce = ws[WS_CE] / (float)N;
    out[0] = 0.5f * ce + 0.5f * proto;   // (1-lambda)*ce + lambda*proto, lambda=0.5
  }
}

extern "C" void kernel_launch(void* const* d_in, const int* in_sizes, int n_in,
                              void* d_out, int out_size, void* d_ws, size_t ws_size,
                              hipStream_t stream) {
  const float* emb    = (const float*)d_in[0];
  const float* logits = (const float*)d_in[1];
  const int*   labels = (const int*)d_in[2];
  float* out = (float*)d_out;
  float* ws  = (float*)d_ws;
  const int N = in_sizes[2];

  const size_t wsBytes = (size_t)(1 + 2 * C_DIM + C_DIM * D_DIM) * sizeof(float);
  hipMemsetAsync(d_ws, 0, wsBytes, stream);

  ce_kernel<<<1024, 256, 0, stream>>>(logits, labels, ws, N);

  proto_kernel<<<256, 256, L_TOTAL, stream>>>(emb, labels, ws, N);

  finalize_kernel<<<1, C_DIM, 0, stream>>>(ws, out, N);
}
